// MultiQueryAttention_32607391711700
// MI455X (gfx1250) — matrix operand order repaired and checked by reading, not compiled, not run on hardware
//
#include <hip/hip_runtime.h>
#include <hip/hip_bf16.h>

typedef __attribute__((ext_vector_type(16))) __bf16 v16bf;
typedef __attribute__((ext_vector_type(8)))  __bf16 v8bf;
typedef __attribute__((ext_vector_type(8)))  float  v8f;

#define NB   2
#define NS   2048
#define DIN  2048
#define DOUT 2048
#define NH   16
#define HD   128
#define KV   64
#define LDQ  1152   /* 1024 (q in kv-space, all heads) + 64 (K) + 64 (V) */
#define CTXD 1024   /* NH*KV */
#define PSTR 40     /* LDS row stride (elements); 80B keeps 16B alignment */

static __device__ __forceinline__ v16bf cat16(v8bf lo, v8bf hi_) {
    return __builtin_shufflevector(lo, hi_, 0, 1, 2, 3, 4, 5, 6, 7,
                                            8, 9, 10, 11, 12, 13, 14, 15);
}

// ---------------- prep kernels ----------------
__global__ void cvt_f32_bf16(const float* __restrict__ in, __bf16* __restrict__ out, int n) {
    int i = blockIdx.x * blockDim.x + threadIdx.x;
    if (i < n) out[i] = (__bf16)in[i];
}

// WcatT[col][d]  (transposed so WMMA B-fragments are contiguous in K)
// col 0:1024   = (Wq[:, h*128:(h+1)*128] @ Wq2kv)[d, k]   (fused 128->64)
// col 1024:1088 = Wk^T, col 1088:1152 = Wv^T
__global__ void fuse_wq(const float* __restrict__ Wq, const float* __restrict__ Wk,
                        const float* __restrict__ Wv, const float* __restrict__ Wq2kv,
                        __bf16* __restrict__ WcatT) {
    int idx = blockIdx.x * 256 + threadIdx.x;
    if (idx >= DIN * LDQ) return;
    int col = idx / DIN, d = idx % DIN;   // consecutive threads -> consecutive d (coalesced store)
    float v;
    if (col < 1024) {
        int h = col >> 6, kk = col & 63;
        const float* wq = Wq + (size_t)d * DOUT + h * HD;
        float s = 0.f;
        for (int j = 0; j < HD; ++j) s += wq[j] * Wq2kv[j * KV + kk];
        v = s;
    } else if (col < 1088) {
        v = Wk[(size_t)d * KV + (col - 1024)];
    } else {
        v = Wv[(size_t)d * KV + (col - 1088)];
    }
    WcatT[(size_t)col * DIN + d] = (__bf16)v;
}

__global__ void fuse_bias_qkv(const float* __restrict__ bq, const float* __restrict__ bk,
                              const float* __restrict__ bv, const float* __restrict__ Wq2kv,
                              float* __restrict__ biascat) {
    int col = blockIdx.x * 256 + threadIdx.x;
    if (col >= LDQ) return;
    float v;
    if (col < 1024) {
        int h = col >> 6, kk = col & 63;
        float s = 0.f;
        for (int j = 0; j < HD; ++j) s += bq[h * HD + j] * Wq2kv[j * KV + kk];
        v = s;
    } else if (col < 1088) v = bk[col - 1024];
    else                   v = bv[col - 1088];
    biascat[col] = v;
}

// WoeffT[n][h*64+k] = sum_j Wkv2h[k,j] * Wo[h*128+j, n]   (fused 64->128->Wo, transposed)
__global__ void fuse_wo(const float* __restrict__ Wkv2h, const float* __restrict__ Wo,
                        __bf16* __restrict__ WoeffT) {
    int idx = blockIdx.x * 256 + threadIdx.x;
    if (idx >= CTXD * DOUT) return;
    int n = idx / CTXD, r = idx % CTXD;   // consecutive threads -> consecutive r (coalesced store)
    int h = r >> 6, kk = r & 63;
    float s = 0.f;
    for (int j = 0; j < HD; ++j)
        s += Wkv2h[kk * HD + j] * Wo[(size_t)(h * HD + j) * DOUT + n];
    WoeffT[(size_t)n * CTXD + r] = (__bf16)s;
}

// bo_eff[n] = bo[n] + sum_h sum_j bkv2h[j] * Wo[h*128+j, n]
__global__ void fuse_bo(const float* __restrict__ bkv2h, const float* __restrict__ Wo,
                        const float* __restrict__ bo, float* __restrict__ boeff) {
    int n = blockIdx.x * 256 + threadIdx.x;
    if (n >= DOUT) return;
    float s = bo[n];
    for (int h = 0; h < NH; ++h)
        for (int j = 0; j < HD; ++j)
            s += bkv2h[j] * Wo[(size_t)(h * HD + j) * DOUT + n];
    boeff[n] = s;
}

// VT[b][n][s] = V[b][s][n]  (so V^T A-fragments are contiguous in key index)
__global__ void transpose_v(const __bf16* __restrict__ qkv, __bf16* __restrict__ vt) {
    int idx = blockIdx.x * 256 + threadIdx.x;
    if (idx >= NB * KV * NS) return;
    int s = idx & (NS - 1);
    int n = (idx >> 11) & (KV - 1);
    int b = idx >> 17;
    vt[idx] = qkv[(size_t)(b * NS + s) * LDQ + 1088 + n];
}

// ---------------- bf16 WMMA GEMM: C[M,N] = A[M,K] @ BT[N,K]^T + bias ----------------
// 4 waves/block; each wave computes a 32x64 tile: 2 A-frags x 4 B-frags -> 8 WMMA / K-step,
// all fragment loads are 128-bit.
template <bool OUT_BF16>
__global__ __launch_bounds__(128)
void gemm_wmma(const __bf16* __restrict__ A, const __bf16* __restrict__ BT,
               const float* __restrict__ bias, void* __restrict__ Cout,
               int M, int N, int K) {
    const int lane = threadIdx.x & 31;
    const int wid  = threadIdx.x >> 5;
    const int m0 = (blockIdx.x * 4 + wid) * 32;
    const int n0 = blockIdx.y * 64;
    const int nlo = lane & 15;
    const int hi  = lane >> 4;

    v8f acc[2][4] = {};
    const v8bf* Ar[2];
    Ar[0] = (const v8bf*)(A + (size_t)(m0 + nlo) * K);
    Ar[1] = (const v8bf*)(A + (size_t)(m0 + 16 + nlo) * K);
    const v8bf* Bc[4];
#pragma unroll
    for (int t = 0; t < 4; ++t)
        Bc[t] = (const v8bf*)(BT + (size_t)(n0 + t * 16 + nlo) * K);

    for (int k0 = 0; k0 < K; k0 += 32) {
        const int kb = k0 >> 3;
        if (k0 + 32 < K) {
            __builtin_prefetch(&Ar[0][kb + 4], 0, 0);   // global_prefetch_b8
            __builtin_prefetch(&Ar[1][kb + 4], 0, 0);
        }
        v16bf a[2], b[4];
#pragma unroll
        for (int r = 0; r < 2; ++r)
            a[r] = cat16(Ar[r][kb + hi], Ar[r][kb + 2 + hi]);
#pragma unroll
        for (int t = 0; t < 4; ++t)
            b[t] = cat16(Bc[t][kb + hi], Bc[t][kb + 2 + hi]);
#pragma unroll
        for (int r = 0; r < 2; ++r)
#pragma unroll
            for (int t = 0; t < 4; ++t)
                acc[r][t] = __builtin_amdgcn_wmma_f32_16x16x32_bf16(
                    false, a[r], false, b[t], (short)0, acc[r][t], false, false);
    }
    const int rbase = hi * 8;
#pragma unroll
    for (int r = 0; r < 2; ++r)
#pragma unroll
        for (int t = 0; t < 4; ++t) {
            const int n = n0 + t * 16 + nlo;
            const float bb = bias[n];
#pragma unroll
            for (int i = 0; i < 8; ++i) {
                const size_t idx = (size_t)(m0 + r * 16 + rbase + i) * N + n;
                float v = acc[r][t][i] + bb;
                if (OUT_BF16) ((__bf16*)Cout)[idx] = (__bf16)v;
                else          ((float*)Cout)[idx]  = v;
            }
        }
}

// ---------------- one 32-key flash-attention block (transposed compute) ----------------
// MASKED=true only for the single diagonal block; all earlier blocks are mask-free.
template <bool MASKED>
static __device__ __forceinline__ void attn_block(
    int c, int q, int rbase, int hi, int nlo,
    const __bf16* __restrict__ qkvrow, const v8bf* const* vrow,
    const v16bf* qb, __bf16* ldsPw,
    float& mrow, float& lrow, v8f* o) {
    // ---- S^T = K @ Q^T  (32 keys x 16 q) ----
    v8f sc[2] = {};
#pragma unroll
    for (int nb = 0; nb < 2; ++nb) {
        const v8bf* kr =
            (const v8bf*)(qkvrow + (size_t)(c + nb * 16 + nlo) * LDQ + 1024);
#pragma unroll
        for (int hf = 0; hf < 2; ++hf) {
            v16bf ka = cat16(kr[hf * 4 + hi], kr[hf * 4 + 2 + hi]);
            sc[nb] = __builtin_amdgcn_wmma_f32_16x16x32_bf16(
                false, ka, false, qb[hf], (short)0, sc[nb], false, false);
        }
    }
    // ---- online softmax: per-lane scalar stats ----
    float s[2][8];
    float vmax = -3.0e38f;
#pragma unroll
    for (int nb = 0; nb < 2; ++nb)
#pragma unroll
        for (int i = 0; i < 8; ++i) {
            float v = sc[nb][i] * 0.125f;             // 1/sqrt(64)
            if (MASKED) {
                if (c + nb * 16 + rbase + i > q) v = -3.0e38f;   // causal mask
            }
            s[nb][i] = v;
            vmax = fmaxf(vmax, v);
        }
    vmax = fmaxf(vmax, __shfl_xor(vmax, 16, 32));     // join partner half-row
    const float mn = fmaxf(mrow, vmax);
    const float al = __expf(mrow - mn);
    mrow = mn;
    float rs = 0.f;
#pragma unroll
    for (int nb = 0; nb < 2; ++nb)
#pragma unroll
        for (int i = 0; i < 8; ++i) {
            const float e = __expf(s[nb][i] - mn);
            s[nb][i] = e;
            rs += e;
        }
    rs += __shfl_xor(rs, 16, 32);
    lrow = lrow * al + rs;
#pragma unroll
    for (int t = 0; t < 4; ++t) o[t] *= al;           // uniform per lane
    // ---- P^T: C-layout -> B-layout via wave-private LDS (ldsP[q][key]) ----
#pragma unroll
    for (int nb = 0; nb < 2; ++nb)
#pragma unroll
        for (int i = 0; i < 8; ++i)
            ldsPw[nlo * PSTR + nb * 16 + rbase + i] = (__bf16)s[nb][i];
    asm volatile("s_wait_dscnt 0" ::: "memory");
    v16bf pb;
    {
        const v8bf* pp = (const v8bf*)&ldsPw[nlo * PSTR];
        pb = cat16(pp[hi], pp[2 + hi]);               // lane's q-column, keys hi*8..+7 and 16+hi*8..+7
    }
    asm volatile("s_wait_dscnt 0" ::: "memory");
    // ---- O^T += V^T @ P^T  (64 kvdim x 16 q) ----
#pragma unroll
    for (int t = 0; t < 4; ++t) {
        const v8bf* vp = vrow[t] + (c >> 3);
        v16bf va = cat16(vp[hi], vp[2 + hi]);
        o[t] = __builtin_amdgcn_wmma_f32_16x16x32_bf16(
            false, va, false, pb, (short)0, o[t], false, false);
    }
}

// ---------------- causal flash attention, computed TRANSPOSED ----------------
// grid: (S/64, B*H), 4 waves/block; each wave owns 16 q rows, streams keys in 32-blocks.
// Main loop is mask-free; exactly one diagonal block per wave gets the masked path.
__global__ __launch_bounds__(128)
void attn_mqa(const __bf16* __restrict__ qkv, const __bf16* __restrict__ vt,
              __bf16* __restrict__ ctx) {
    __shared__ alignas(16) __bf16 ldsP[4][16 * PSTR];
    const int lane = threadIdx.x & 31;
    const int wid  = threadIdx.x >> 5;
    const int b = blockIdx.y >> 4;
    const int h = blockIdx.y & 15;
    const int qs = (blockIdx.x * 4 + wid) * 16;
    const int nlo = lane & 15;
    const int hi  = lane >> 4;
    const int rbase = hi * 8;
    const size_t rowbase = (size_t)b * NS * LDQ;
    const int q = qs + nlo;
    const __bf16* qkvrow = qkv + rowbase;
    __bf16* ldsPw = &ldsP[wid][0];

    // Q^T B-fragments (64 kvdim x 16 q -> two 32x16 halves), 128-bit loads
    v16bf qb[2];
    {
        const v8bf* qp = (const v8bf*)(qkvrow + (size_t)(qs + nlo) * LDQ + h * KV);
#pragma unroll
        for (int hf = 0; hf < 2; ++hf)
            qb[hf] = cat16(qp[hf * 4 + hi], qp[hf * 4 + 2 + hi]);
    }
    // V^T A-fragment row pointers (VT rows are contiguous in key)
    const v8bf* vrow[4];
#pragma unroll
    for (int t = 0; t < 4; ++t)
        vrow[t] = (const v8bf*)(vt + (size_t)(b * KV + t * 16 + nlo) * NS);

    float mrow = -3.0e38f, lrow = 0.f;
    v8f o[4] = {};

    // Diagonal block start: the only block needing the causal mask.
    const int clast = (qs + 15) & ~31;               // keys [clast, clast+32) straddle diag
    for (int c = 0; c < clast; c += 32)
        attn_block<false>(c, q, rbase, hi, nlo, qkvrow, vrow, qb, ldsPw, mrow, lrow, o);
    attn_block<true>(clast, q, rbase, hi, nlo, qkvrow, vrow, qb, ldsPw, mrow, lrow, o);

    // ---- normalize + store ctx: 8 contiguous kv-dims per t -> b128 stores ----
    const float inv_l = 1.f / lrow;
#pragma unroll
    for (int t = 0; t < 4; ++t) {
        v8bf ov;
#pragma unroll
        for (int i = 0; i < 8; ++i) ov[i] = (__bf16)(o[t][i] * inv_l);
        *(v8bf*)&ctx[(size_t)(b * NS + q) * CTXD + h * KV + t * 16 + rbase] = ov;
    }
}

// ---------------- launch ----------------
extern "C" void kernel_launch(void* const* d_in, const int* in_sizes, int n_in,
                              void* d_out, int out_size, void* d_ws, size_t ws_size,
                              hipStream_t stream) {
    const float* x     = (const float*)d_in[0];
    const float* Wq    = (const float*)d_in[1];
    const float* bq    = (const float*)d_in[2];
    const float* Wk    = (const float*)d_in[3];
    const float* bk    = (const float*)d_in[4];
    const float* Wv    = (const float*)d_in[5];
    const float* bv    = (const float*)d_in[6];
    const float* Wq2kv = (const float*)d_in[7];
    const float* Wkv2h = (const float*)d_in[8];
    const float* bkv2h = (const float*)d_in[9];
    const float* Wo    = (const float*)d_in[10];
    const float* bo    = (const float*)d_in[11];

    char* ws = (char*)d_ws;
    __bf16* xb      = (__bf16*)(ws + 0);          // 16,777,216 B
    __bf16* wcatT   = (__bf16*)(ws + 16777216);   //  4,718,592 B
    float*  biascat = (float*) (ws + 21495808);   //      4,608 B
    __bf16* qkv     = (__bf16*)(ws + 21500416);   //  9,437,184 B
    __bf16* ctx     = (__bf16*)(ws + 30937600);   //  8,388,608 B
    __bf16* woeffT  = (__bf16*)(ws + 39326208);   //  4,194,304 B
    float*  boeff   = (float*) (ws + 43520512);   //      8,192 B
    __bf16* vtr     = (__bf16*)(ws + 43528704);   //    524,288 B

    const int NX = NB * NS * DIN;  // 8,388,608
    cvt_f32_bf16<<<(NX + 255) / 256, 256, 0, stream>>>(x, xb, NX);
    fuse_wq<<<(DIN * LDQ + 255) / 256, 256, 0, stream>>>(Wq, Wk, Wv, Wq2kv, wcatT);
    fuse_bias_qkv<<<(LDQ + 255) / 256, 256, 0, stream>>>(bq, bk, bv, Wq2kv, biascat);
    fuse_wo<<<(CTXD * DOUT + 255) / 256, 256, 0, stream>>>(Wkv2h, Wo, woeffT);
    fuse_bo<<<(DOUT + 255) / 256, 256, 0, stream>>>(bkv2h, Wo, bo, boeff);

    // qkv = x @ [Wq_eff | Wk | Wv] + bias   : [4096,2048] @ [2048,1152]
    gemm_wmma<true><<<dim3((NB * NS) / 128, LDQ / 64), 128, 0, stream>>>(
        xb, wcatT, biascat, (void*)qkv, NB * NS, LDQ, DIN);

    transpose_v<<<(NB * KV * NS + 255) / 256, 256, 0, stream>>>(qkv, vtr);

    // causal flash attention per (b,h), 64 q-rows per block
    attn_mqa<<<dim3(NS / 64, NB * NH), 128, 0, stream>>>(qkv, vtr, ctx);

    // out = ctx @ Wo_eff + bo_eff          : [4096,1024] @ [1024,2048] -> f32
    gemm_wmma<false><<<dim3((NB * NS) / 128, DOUT / 64), 128, 0, stream>>>(
        ctx, woeffT, boeff, d_out, NB * NS, DOUT, CTXD);
}
